// CLAMSB_88459146428743
// MI455X (gfx1250) — matrix operand order repaired and checked by reading, not compiled, not run on hardware
//
#include <hip/hip_runtime.h>

#define NN     65536
#define D_IN   1024
#define D1     512
#define D2     256
#define CLS    2
#define KK     8

typedef __attribute__((ext_vector_type(16))) __bf16 v16bf;
typedef __attribute__((ext_vector_type(8)))  float  v8f;

#define NEG_INF (-3.4028235e38f)
#define POS_INF ( 3.4028235e38f)

union Frag { uint4 u[2]; v16bf v; };

// Native hardware RNE conversion (v_cvt_pk_bf16_f32 path).
__device__ __forceinline__ __bf16 f2bf(float f) { return (__bf16)f; }
__device__ __forceinline__ unsigned short f2bfu(float f) {
    return __builtin_bit_cast(unsigned short, (__bf16)f);
}
__device__ __forceinline__ float bfu2f(unsigned short s) {
    return __builtin_bit_cast(float, ((unsigned)s) << 16);
}

// ---------------------------------------------------------------------------
// Kernel 0: one-time weight conversion + transpose to bf16, K-contiguous per
// column (exactly the WMMA B-fragment layout). All three stay L2-resident.
//   W1  [1024][512] -> W1t [512][1024]
//   Wa  [512][256]  -> Wat [256][512]
//   Wb  [512][256]  -> Wbt [256][512]
// ---------------------------------------------------------------------------
__global__ __launch_bounds__(256) void k_prep(const float* __restrict__ W1,
                                              const float* __restrict__ Wa,
                                              const float* __restrict__ Wb,
                                              unsigned short* __restrict__ W1t,
                                              unsigned short* __restrict__ Wat,
                                              unsigned short* __restrict__ Wbt) {
    const int stride = gridDim.x * blockDim.x;
    const int id = blockIdx.x * blockDim.x + threadIdx.x;
    for (int e = id; e < D_IN * D1; e += stride) {
        int k = e >> 9, c = e & (D1 - 1);          // coalesced read over c
        W1t[(size_t)c * D_IN + k] = f2bfu(W1[e]);
    }
    for (int e = id; e < D1 * D2; e += stride) {
        int k = e >> 8, c = e & (D2 - 1);
        Wat[(size_t)c * D1 + k] = f2bfu(Wa[e]);
        Wbt[(size_t)c * D1 + k] = f2bfu(Wb[e]);
    }
}

// ---------------------------------------------------------------------------
// Kernel 1: h = relu(x @ W1 + b1)   [65536,1024]x[1024,512] -> bf16 h
// Block tile 32(M) x 256(N), 8 waves as 2x4, wave tile 16x64.
// A-fragments: fp32 x from HBM (HW cvt). B-fragments: bf16 W1t from L2.
// All 4 B fragments of a k-step are preloaded before the WMMA chain so the
// loads clause together and overlap the matrix pipe.
// ---------------------------------------------------------------------------
__global__ __launch_bounds__(256) void k_gemm1(const float* __restrict__ x,
                                               const unsigned short* __restrict__ W1t,
                                               const float* __restrict__ b1,
                                               unsigned short* __restrict__ hout) {
    const int t  = threadIdx.x;
    const int l  = t & 31;
    const int w  = t >> 5;
    const int wm = w >> 2;        // 0..1
    const int wn = w & 3;         // 0..3
    const int r0 = blockIdx.x * 32 + wm * 16;
    const int c0 = blockIdx.y * 256 + wn * 64;

    v8f acc[4] = {};

    const int row   = r0 + (l & 15);
    const int klo   = (l < 16) ? 0 : 8;       // A: interleaved K halves
    const int khalf = (l >> 4) << 3;          // B: interleaved K halves (0 or 8)

    const unsigned short* bbase[4];
    #pragma unroll
    for (int tt = 0; tt < 4; ++tt)
        bbase[tt] = W1t + (size_t)(c0 + tt * 16 + (l & 15)) * D_IN + khalf;

    const float* ap = x + (size_t)row * D_IN + klo;

    for (int k0 = 0; k0 < D_IN; k0 += 32) {
        // A tile slice: 16 fp32 -> bf16 (HW pack-convert).
        float4 f0 = *(const float4*)(ap + k0);
        float4 f1 = *(const float4*)(ap + k0 + 4);
        float4 f2 = *(const float4*)(ap + k0 + 16);
        float4 f3 = *(const float4*)(ap + k0 + 20);

        // Preload all B fragments (one clause of 8x b128 loads).
        Frag bf[4];
        #pragma unroll
        for (int tt = 0; tt < 4; ++tt) {
            const uint4* bp = (const uint4*)(bbase[tt] + k0);
            bf[tt].u[0] = bp[0];
            bf[tt].u[1] = bp[2];
        }

        v16bf a;
        a[0]  = f2bf(f0.x); a[1]  = f2bf(f0.y); a[2]  = f2bf(f0.z); a[3]  = f2bf(f0.w);
        a[4]  = f2bf(f1.x); a[5]  = f2bf(f1.y); a[6]  = f2bf(f1.z); a[7]  = f2bf(f1.w);
        a[8]  = f2bf(f2.x); a[9]  = f2bf(f2.y); a[10] = f2bf(f2.z); a[11] = f2bf(f2.w);
        a[12] = f2bf(f3.x); a[13] = f2bf(f3.y); a[14] = f2bf(f3.z); a[15] = f2bf(f3.w);

        #pragma unroll
        for (int tt = 0; tt < 4; ++tt)
            acc[tt] = __builtin_amdgcn_wmma_f32_16x16x32_bf16(
                false, a, false, bf[tt].v, (short)0, acc[tt], false, false);
    }

    // Epilogue: bias + ReLU + bf16 store.
    const int rbase = r0 + ((l >> 4) << 3);
    #pragma unroll
    for (int tt = 0; tt < 4; ++tt) {
        int colg = c0 + tt * 16 + (l & 15);
        float bias = b1[colg];
        #pragma unroll
        for (int i = 0; i < 8; ++i) {
            float v = acc[tt][i] + bias;
            v = v > 0.f ? v : 0.f;
            hout[(size_t)(rbase + i) * D1 + colg] = f2bfu(v);
        }
    }
}

// ---------------------------------------------------------------------------
// Kernel 2: attn_raw[c][n] = sum_j tanh(hWa+ba)*sigmoid(hWb+bb) * Wc[j][c] + bc
// Block tile 32 rows x all 256 D2 cols; two WMMA chains per tile.
// h (bf16, L2-resident) feeds A directly; Wat/Wbt feed B directly, with all
// 8 B fragments preloaded per k-step.
// ---------------------------------------------------------------------------
__global__ __launch_bounds__(256) void k_attn(const unsigned short* __restrict__ h,
                                              const unsigned short* __restrict__ Wat,
                                              const unsigned short* __restrict__ Wbt,
                                              const float* __restrict__ ba,
                                              const float* __restrict__ bbias,
                                              const float* __restrict__ Wc,
                                              const float* __restrict__ bc,
                                              float* __restrict__ attn_raw) {
    __shared__ float red[2][32][4];

    const int t  = threadIdx.x;
    const int l  = t & 31;
    const int w  = t >> 5;
    const int wm = w >> 2;
    const int wn = w & 3;
    const int r0 = blockIdx.x * 32 + wm * 16;
    const int n0 = wn * 64;

    v8f acca[4] = {};
    v8f accg[4] = {};

    const int row   = r0 + (l & 15);
    const int klo   = (l < 16) ? 0 : 16;
    const int khalf = (l >> 4) << 4;

    const unsigned short* abase = h + (size_t)row * D1 + klo;
    const unsigned short* wabase[4];
    const unsigned short* wbbase[4];
    #pragma unroll
    for (int tt = 0; tt < 4; ++tt) {
        int colg = n0 + tt * 16 + (l & 15);
        wabase[tt] = Wat + (size_t)colg * D1 + khalf;
        wbbase[tt] = Wbt + (size_t)colg * D1 + khalf;
    }

    for (int k0 = 0; k0 < D1; k0 += 32) {
        const uint4* ap = (const uint4*)(abase + k0);
        Frag af;
        af.u[0] = ap[0];
        af.u[1] = ap[1];   // +8 bf16 elements

        Frag bfa[4], bfb[4];
        #pragma unroll
        for (int tt = 0; tt < 4; ++tt) {
            const uint4* bpa = (const uint4*)(wabase[tt] + k0);
            const uint4* bpb = (const uint4*)(wbbase[tt] + k0);
            bfa[tt].u[0] = bpa[0]; bfa[tt].u[1] = bpa[1];
            bfb[tt].u[0] = bpb[0]; bfb[tt].u[1] = bpb[1];
        }

        #pragma unroll
        for (int tt = 0; tt < 4; ++tt) {
            acca[tt] = __builtin_amdgcn_wmma_f32_16x16x32_bf16(
                false, af.v, false, bfa[tt].v, (short)0, acca[tt], false, false);
            accg[tt] = __builtin_amdgcn_wmma_f32_16x16x32_bf16(
                false, af.v, false, bfb[tt].v, (short)0, accg[tt], false, false);
        }
    }

    // Fused tanh * sigmoid, then contract with Wc (2 classes).
    float part[8][2];
    #pragma unroll
    for (int i = 0; i < 8; ++i) { part[i][0] = 0.f; part[i][1] = 0.f; }

    #pragma unroll
    for (int tt = 0; tt < 4; ++tt) {
        int colg = n0 + tt * 16 + (l & 15);
        float wc0 = Wc[colg * 2 + 0];
        float wc1 = Wc[colg * 2 + 1];
        float bav = ba[colg];
        float bbv = bbias[colg];
        #pragma unroll
        for (int i = 0; i < 8; ++i) {
            float av = tanhf(acca[tt][i] + bav);
            float gv = 1.f / (1.f + expf(-(accg[tt][i] + bbv)));
            float ag = av * gv;
            part[i][0] += ag * wc0;
            part[i][1] += ag * wc1;
        }
    }
    // Reduce over the 16 lanes sharing a row group.
    #pragma unroll
    for (int mm = 8; mm >= 1; mm >>= 1) {
        #pragma unroll
        for (int i = 0; i < 8; ++i) {
            part[i][0] += __shfl_xor(part[i][0], mm, 32);
            part[i][1] += __shfl_xor(part[i][1], mm, 32);
        }
    }
    if ((l & 15) == 0) {
        int rb = wm * 16 + ((l >> 4) << 3);
        #pragma unroll
        for (int i = 0; i < 8; ++i) {
            red[0][rb + i][wn] = part[i][0];
            red[1][rb + i][wn] = part[i][1];
        }
    }
    __syncthreads();
    if (t < 64) {
        int c = t >> 5, rr = t & 31;
        float s = red[c][rr][0] + red[c][rr][1] + red[c][rr][2] + red[c][rr][3] + bc[c];
        attn_raw[(size_t)c * NN + blockIdx.x * 32 + rr] = s;
    }
}

// ---------------------------------------------------------------------------
// Kernel 3a: per-block online (max, sum_exp) for both classes.
// ---------------------------------------------------------------------------
__global__ void k_reduce1(const float* __restrict__ raw, float* __restrict__ partials) {
    __shared__ float sm[256], ss[256];
    const int t = threadIdx.x;
    const int n = blockIdx.x * 256 + t;
    for (int c = 0; c < 2; ++c) {
        float v = raw[(size_t)c * NN + n];
        sm[t] = v; ss[t] = 1.0f;
        __syncthreads();
        for (int off = 128; off > 0; off >>= 1) {
            if (t < off) {
                float m2 = sm[t + off], s2 = ss[t + off];
                float M  = fmaxf(sm[t], m2);
                ss[t] = ss[t] * expf(sm[t] - M) + s2 * expf(m2 - M);
                sm[t] = M;
            }
            __syncthreads();
        }
        if (t == 0) {
            partials[blockIdx.x * 4 + c * 2 + 0] = sm[0];
            partials[blockIdx.x * 4 + c * 2 + 1] = ss[0];
        }
        __syncthreads();
    }
}

// ---------------------------------------------------------------------------
// Kernel 3b: combine 256 partials -> gstat; zero M (ws is poisoned).
// ---------------------------------------------------------------------------
__global__ void k_reduce2(const float* __restrict__ partials,
                          float* __restrict__ gstat, float* __restrict__ Mbuf) {
    __shared__ float sm[256], ss[256];
    const int t = threadIdx.x;
    for (int c = 0; c < 2; ++c) {
        sm[t] = partials[t * 4 + c * 2 + 0];
        ss[t] = partials[t * 4 + c * 2 + 1];
        __syncthreads();
        for (int off = 128; off > 0; off >>= 1) {
            if (t < off) {
                float m2 = sm[t + off], s2 = ss[t + off];
                float M  = fmaxf(sm[t], m2);
                ss[t] = ss[t] * expf(sm[t] - M) + s2 * expf(m2 - M);
                sm[t] = M;
            }
            __syncthreads();
        }
        if (t == 0) { gstat[c * 2] = sm[0]; gstat[c * 2 + 1] = ss[0]; }
        __syncthreads();
    }
    Mbuf[t] = 0.f; Mbuf[t + 256] = 0.f; Mbuf[t + 512] = 0.f; Mbuf[t + 768] = 0.f;
}

// ---------------------------------------------------------------------------
// Kernel 4: M[c] = sum_n softmax(raw)[c][n] * h[n]   (h in L2, coalesced)
// ---------------------------------------------------------------------------
__global__ __launch_bounds__(256) void k_slidefeat(const float* __restrict__ raw,
                                                   const unsigned short* __restrict__ h,
                                                   const float* __restrict__ gstat,
                                                   float* __restrict__ Mbuf) {
    __shared__ float wl[2][1024];
    const int t  = threadIdx.x;
    const int n0 = blockIdx.x * 1024;
    float m0 = gstat[0], s0 = gstat[1], m1 = gstat[2], s1 = gstat[3];
    #pragma unroll
    for (int i = 0; i < 4; ++i) {
        int j = t + i * 256;
        wl[0][j] = expf(raw[(size_t)0 * NN + n0 + j] - m0) / s0;
        wl[1][j] = expf(raw[(size_t)1 * NN + n0 + j] - m1) / s1;
    }
    __syncthreads();
    const int d0 = t * 2;
    float a00 = 0.f, a01 = 0.f, a10 = 0.f, a11 = 0.f;
    for (int j = 0; j < 1024; ++j) {
        unsigned u = *(const unsigned*)(h + (size_t)(n0 + j) * D1 + d0);
        float flo = __builtin_bit_cast(float, u << 16);
        float fhi = __builtin_bit_cast(float, u & 0xffff0000u);
        float w0 = wl[0][j], w1 = wl[1][j];
        a00 += w0 * flo; a01 += w0 * fhi;
        a10 += w1 * flo; a11 += w1 * fhi;
    }
    atomicAdd(&Mbuf[d0 + 0], a00);
    atomicAdd(&Mbuf[d0 + 1], a01);
    atomicAdd(&Mbuf[512 + d0 + 0], a10);
    atomicAdd(&Mbuf[512 + d0 + 1], a11);
}

// ---------------------------------------------------------------------------
// Kernel 5: per-class top-8 / bottom-8 indices of attn_raw (ties -> lower idx)
// ---------------------------------------------------------------------------
__global__ __launch_bounds__(256) void k_topk(const float* __restrict__ raw_all,
                                              int* __restrict__ topi, int* __restrict__ boti) {
    const int c = blockIdx.x;
    const int t = threadIdx.x;
    const float* raw = raw_all + (size_t)c * NN;

    float tv[KK]; int ti[KK];
    float bv[KK]; int bi[KK];
    #pragma unroll
    for (int j = 0; j < KK; ++j) { tv[j] = NEG_INF; ti[j] = 0x7fffffff; bv[j] = POS_INF; bi[j] = 0x7fffffff; }

    for (int n = t; n < NN; n += 256) {
        float v = raw[n];
        if (v > tv[KK - 1]) {
            int j = KK - 1;
            while (j > 0 && v > tv[j - 1]) { tv[j] = tv[j - 1]; ti[j] = ti[j - 1]; --j; }
            tv[j] = v; ti[j] = n;
        }
        if (v < bv[KK - 1]) {
            int j = KK - 1;
            while (j > 0 && v < bv[j - 1]) { bv[j] = bv[j - 1]; bi[j] = bi[j - 1]; --j; }
            bv[j] = v; bi[j] = n;
        }
    }

    __shared__ float cv[256];
    __shared__ int   ci[256];
    __shared__ int   ct[256];
    __shared__ int   winner;

    // Top-8 extraction (argmax).
    int pos = 0;
    for (int round = 0; round < KK; ++round) {
        cv[t] = (pos < KK) ? tv[pos] : NEG_INF;
        ci[t] = (pos < KK) ? ti[pos] : 0x7fffffff;
        ct[t] = t;
        __syncthreads();
        for (int off = 128; off > 0; off >>= 1) {
            if (t < off) {
                bool take = (cv[t + off] > cv[t]) ||
                            (cv[t + off] == cv[t] && ci[t + off] < ci[t]);
                if (take) { cv[t] = cv[t + off]; ci[t] = ci[t + off]; ct[t] = ct[t + off]; }
            }
            __syncthreads();
        }
        if (t == 0) { topi[c * KK + round] = ci[0]; winner = ct[0]; }
        __syncthreads();
        if (t == winner) ++pos;
        __syncthreads();
    }
    // Bottom-8 extraction (argmin).
    pos = 0;
    for (int round = 0; round < KK; ++round) {
        cv[t] = (pos < KK) ? bv[pos] : POS_INF;
        ci[t] = (pos < KK) ? bi[pos] : 0x7fffffff;
        ct[t] = t;
        __syncthreads();
        for (int off = 128; off > 0; off >>= 1) {
            if (t < off) {
                bool take = (cv[t + off] < cv[t]) ||
                            (cv[t + off] == cv[t] && ci[t + off] < ci[t]);
                if (take) { cv[t] = cv[t + off]; ci[t] = ci[t + off]; ct[t] = ct[t + off]; }
            }
            __syncthreads();
        }
        if (t == 0) { boti[c * KK + round] = ci[0]; winner = ct[0]; }
        __syncthreads();
        if (t == winner) ++pos;
        __syncthreads();
    }
}

// ---------------------------------------------------------------------------
// Kernel 6: instance-eval CE losses + slide logits / softmax / argmax.
// ---------------------------------------------------------------------------
__global__ __launch_bounds__(256) void k_final(const unsigned short* __restrict__ h,
                                               const float* __restrict__ Mbuf,
                                               const int* __restrict__ topi,
                                               const int* __restrict__ boti,
                                               const int* __restrict__ target,
                                               const float* __restrict__ Wins,
                                               const float* __restrict__ bins,
                                               const float* __restrict__ Wcls,
                                               const float* __restrict__ bcls,
                                               float* __restrict__ out) {
    __shared__ float lp[32][2];
    __shared__ float slide[2];
    const int t = threadIdx.x, lane = t & 31, wv = t >> 5;

    for (int r = 0; r < 5; ++r) {
        int job = r * 8 + wv;
        if (job < 32) {
            int c = job >> 4, slot = job & 15;
            int idx = (slot < KK) ? topi[c * KK + slot] : boti[c * KK + slot - KK];
            float p0 = 0.f, p1 = 0.f;
            for (int d = lane; d < D1; d += 32) {
                float v = bfu2f(h[(size_t)idx * D1 + d]);
                p0 += v * Wins[c * (D1 * 2) + d * 2 + 0];
                p1 += v * Wins[c * (D1 * 2) + d * 2 + 1];
            }
            #pragma unroll
            for (int mm = 16; mm >= 1; mm >>= 1) {
                p0 += __shfl_xor(p0, mm, 32);
                p1 += __shfl_xor(p1, mm, 32);
            }
            if (lane == 0) {
                float l0 = p0 + bins[c * 2 + 0];
                float l1 = p1 + bins[c * 2 + 1];
                float mx = fmaxf(l0, l1);
                float lse = mx + logf(expf(l0 - mx) + expf(l1 - mx));
                lp[job][0] = l0 - lse;
                lp[job][1] = l1 - lse;
            }
        } else if (job < 34) {
            int c = job - 32;
            float p = 0.f;
            for (int d = lane; d < D1; d += 32) p += Mbuf[c * D1 + d] * Wcls[d];
            #pragma unroll
            for (int mm = 16; mm >= 1; mm >>= 1) p += __shfl_xor(p, mm, 32);
            if (lane == 0) slide[c] = p + bcls[0];
        }
    }
    __syncthreads();
    if (t == 0) {
        int tgt = target[0];
        float acc = 0.f;
        for (int c = 0; c < 2; ++c) {
            float s = 0.f, loss;
            if (c == tgt) {
                for (int j = 0; j < KK; ++j)       s += lp[c * 16 + j][1];
                for (int j = KK; j < 2 * KK; ++j)  s += lp[c * 16 + j][0];
                loss = -s / (2.f * KK);
            } else {
                for (int j = 0; j < KK; ++j)       s += lp[c * 16 + j][0];
                loss = -s / (float)KK;
            }
            acc += loss;
        }
        float l0 = slide[0], l1 = slide[1];
        float mx = fmaxf(l0, l1);
        float e0 = expf(l0 - mx), e1 = expf(l1 - mx);
        float Z = e0 + e1;
        out[0] = l0;
        out[1] = l1;
        out[2] = e0 / Z;
        out[3] = e1 / Z;
        out[4] = (l1 > l0) ? 1.0f : 0.0f;   // argmax, ties -> first
        out[5 + 2 * NN] = acc * 0.5f;       // inst_loss
    }
}

// ---------------------------------------------------------------------------
extern "C" void kernel_launch(void* const* d_in, const int* in_sizes, int n_in,
                              void* d_out, int out_size, void* d_ws, size_t ws_size,
                              hipStream_t stream) {
    const float* x      = (const float*)d_in[0];
    const int*   target = (const int*)  d_in[1];
    const float* W1     = (const float*)d_in[2];
    const float* b1     = (const float*)d_in[3];
    const float* Wa     = (const float*)d_in[4];
    const float* ba     = (const float*)d_in[5];
    const float* Wb     = (const float*)d_in[6];
    const float* bbv    = (const float*)d_in[7];
    const float* Wc     = (const float*)d_in[8];
    const float* bc     = (const float*)d_in[9];
    const float* Wcls   = (const float*)d_in[10];
    const float* bcls   = (const float*)d_in[11];
    const float* Wins   = (const float*)d_in[12];
    const float* bins   = (const float*)d_in[13];

    float* out = (float*)d_out;
    float* attn_raw = out + 5;                // [2, 65536] region of d_out

    char* ws = (char*)d_ws;
    unsigned short* h    = (unsigned short*)ws;                // 64 MB bf16
    float* partials      = (float*)(ws + 67108864);            // 256*4 floats
    float* gstat         = (float*)(ws + 67112960);            // 4 floats
    float* Mbuf          = (float*)(ws + 67113024);            // 2*512 floats
    int*   topi          = (int*)  (ws + 67117120);            // 2*8
    int*   boti          = (int*)  (ws + 67117184);            // 2*8
    unsigned short* W1t  = (unsigned short*)(ws + 67117248);   // 512*1024 bf16
    unsigned short* Wat  = (unsigned short*)(ws + 68165824);   // 256*512 bf16
    unsigned short* Wbt  = (unsigned short*)(ws + 68427968);   // 256*512 bf16

    k_prep     <<<512, 256, 0, stream>>>(W1, Wa, Wb, W1t, Wat, Wbt);
    k_gemm1    <<<dim3(NN / 32, D1 / 256), 256, 0, stream>>>(x, W1t, b1, h);
    k_attn     <<<NN / 32, 256, 0, stream>>>(h, Wat, Wbt, ba, bbv, Wc, bc, attn_raw);
    k_reduce1  <<<NN / 256, 256, 0, stream>>>(attn_raw, partials);
    k_reduce2  <<<1, 256, 0, stream>>>(partials, gstat, Mbuf);
    k_slidefeat<<<NN / 1024, 256, 0, stream>>>(attn_raw, h, gstat, Mbuf);
    k_topk     <<<CLS, 256, 0, stream>>>(attn_raw, topi, boti);
    k_final    <<<1, 256, 0, stream>>>(h, Mbuf, topi, boti, target,
                                       Wins, bins, Wcls, bcls, out);
}
